// S4Model_82231443849496
// MI455X (gfx1250) — compile-verified, matrix operand order checked by reading, not run on hardware
//
#include <hip/hip_runtime.h>
#include <hip/hip_bf16.h>

// ---------------- problem constants (match reference) ----------------
#define Bsz 128
#define Lsz 512
#define Hsz 256
#define Nsz 64
#define NLayers 4
#define LB (Lsz * Bsz)   // 65536 columns; canonical column index jc = b*Lsz + l

typedef __attribute__((ext_vector_type(16))) __bf16 v16bf;
typedef __attribute__((ext_vector_type(8)))  float  v8f;
typedef __attribute__((ext_vector_type(4)))  unsigned int v4u;
typedef __attribute__((ext_vector_type(8)))  int v8i;
typedef __attribute__((ext_vector_type(4)))  int v4i;

// D = A(16x32 bf16) x B(32x16 bf16) + C(16x16 f32), wave32 WMMA.
__device__ __forceinline__ v8f wmma_bf16(v16bf a, v16bf b, v8f c) {
  return __builtin_amdgcn_wmma_f32_16x16x32_bf16(
      /*neg_a=*/false, a, /*neg_b=*/false, b,
      /*c_mod=*/(short)0, c, /*reuse_a=*/false, /*reuse_b=*/false);
}

// K index for element e (0..15) of a 16-bit A/B fragment (ISA 7.12.2):
// elements 0..7 -> K = (hi?8:0)+0..7 ; elements 8..15 -> K = 16+(hi?8:0)+0..7
// (two contiguous 8-runs per lane -> compiler merges to b128 loads)
__device__ __forceinline__ int frag_k(int e, int hi) {
  int i = e >> 1;
  int base = (i < 4) ? (2 * i) : (16 + 2 * (i - 4));
  return base + (hi ? 8 : 0) + (e & 1);
}

__device__ __forceinline__ float gelu_tanh(float v) {
  float v3 = v * v * v;
  return 0.5f * v * (1.f + tanhf(0.7978845608f * (v + 0.044715f * v3)));
}

// ---------------- weight pre-conversion f32 -> bf16 ----------------
__global__ void wconv_kernel(const float* __restrict__ enc_w,
                             const float* __restrict__ out_w,
                             __bf16* __restrict__ Wencb,
                             __bf16* __restrict__ Woutb) {
  int i = blockIdx.x * blockDim.x + threadIdx.x;
  if (i < Hsz * Hsz) Wencb[i] = (__bf16)enc_w[i];
  int tot = NLayers * 2 * Hsz * Hsz;
  for (int t = i; t < tot; t += gridDim.x * blockDim.x)
    Woutb[t] = (__bf16)out_w[t];
}

// ---------------- S4D kernel generation: k0,k1 (H x L) ----------------
__global__ void kgen_kernel(const float* __restrict__ log_dt,
                            const float* __restrict__ log_A_real,
                            const float* __restrict__ A_imag,
                            const float* __restrict__ Cr,
                            const float* __restrict__ Ci,
                            int layer,
                            float* __restrict__ k0,
                            float* __restrict__ k1) {
  int h = blockIdx.x;            // H blocks
  int l = threadIdx.x;           // L threads
  float dt = __expf(log_dt[layer * Hsz + h]);
  float acc0 = 0.f, acc1 = 0.f;
  for (int n = 0; n < Nsz; ++n) {
    int idx = (layer * Hsz + h) * Nsz + n;
    float Ar = -__expf(log_A_real[idx]);
    float Ai = A_imag[idx];
    float wr = Ar * dt, wi = Ai * dt;
    float ew  = __expf(wr);                    // em1 = exp(dtA) - 1
    float e_r = ew * __cosf(wi) - 1.f;
    float e_i = ew * __sinf(wi);
    float den = Ar * Ar + Ai * Ai;             // q = em1 / A
    float qr = (e_r * Ar + e_i * Ai) / den;
    float qi = (e_i * Ar - e_r * Ai) / den;
    float mag = __expf(wr * (float)l);         // vand = exp(dtA*l)
    float ang = wi * (float)l;
    float pr = mag * __cosf(ang);
    float pi = mag * __sinf(ang);
    float sr = qr * pr - qi * pi;              // s = q * vand
    float si = qr * pi + qi * pr;
    float cr0 = Cr[((layer * 2 + 0) * Hsz + h) * Nsz + n];
    float ci0 = Ci[((layer * 2 + 0) * Hsz + h) * Nsz + n];
    float cr1 = Cr[((layer * 2 + 1) * Hsz + h) * Nsz + n];
    float ci1 = Ci[((layer * 2 + 1) * Hsz + h) * Nsz + n];
    acc0 += cr0 * sr - ci0 * si;               // Re(C*s)
    acc1 += cr1 * sr - ci1 * si;
  }
  k0[h * Lsz + l] = 2.f * acc0;
  k1[h * Lsz + l] = 2.f * acc1;
}

// ---------------- encoder: U(H x LB, jc=b*L+l) = enc_w @ X^T + enc_b ----------
// x row for column jc is x + jc*H  (since jc = b*L + l and x is (B,L,H))
__global__ void encoder_kernel(const float* __restrict__ x,
                               const __bf16* __restrict__ Wb,
                               const float* __restrict__ enc_b,
                               float* __restrict__ U,
                               __bf16* __restrict__ Ub) {
  int wave = threadIdx.x >> 5;
  int lane = threadIdx.x & 31;
  int tile = blockIdx.x * 4 + wave;     // 16 * 4096 tiles
  int jt = tile & 4095;                 // LB/16
  int ot = tile >> 12;                  // H/16
  int hi = lane >> 4;
  int nm = lane & 15;
  int jc = jt * 16 + nm;
  const float* xrow = x + (size_t)jc * Hsz;
  v8f c = {};
  for (int kk = 0; kk < Hsz; kk += 32) {
    v16bf a, bb;
#pragma unroll
    for (int e = 0; e < 16; ++e) {
      int k = frag_k(e, hi);
      a[e]  = Wb[(size_t)(ot * 16 + nm) * Hsz + kk + k];
      bb[e] = (__bf16)xrow[kk + k];
    }
    c = wmma_bf16(a, bb, c);
  }
#pragma unroll
  for (int r = 0; r < 8; ++r) {
    int o = ot * 16 + r + hi * 8;       // C/D: M = r (+8 for hi lanes), N = nm
    float v = c[r] + enc_b[o];
    U [(size_t)o * LB + jc] = v;
    Ub[(size_t)o * LB + jc] = (__bf16)v;
  }
}

// ---------------- bidirectional Toeplitz conv + D-skip + GELU ----------------
// Per h: Y(L x B) = T_h(L x L) @ U_h,  T_h[l,m] = (m<=l) ? k0[l-m] : k1[m-l-1]
// (H,B,L) layout: B-fragment K-runs are contiguous in memory (b128 loads),
// C/D epilogue per lane is 8 consecutive l's at fixed b (vectorized I/O).
__global__ void conv_kernel(const __bf16* __restrict__ Ub,
                            const float* __restrict__ U,
                            const float* __restrict__ k0f,
                            const float* __restrict__ k1f,
                            const float* __restrict__ Dp,
                            int layer,
                            __bf16* __restrict__ Yact) {
  __shared__ __bf16 sk0[Lsz];
  __shared__ __bf16 sk1[Lsz];
  int h  = blockIdx.x >> 5;             // H
  int lt = blockIdx.x & 31;             // L/16
  for (int t = threadIdx.x; t < Lsz; t += blockDim.x) {
    sk0[t] = (__bf16)k0f[h * Lsz + t];
    sk1[t] = (__bf16)k1f[h * Lsz + t];
  }
  __syncthreads();

  int wave = threadIdx.x >> 5;
  int lane = threadIdx.x & 31;
  int hi = lane >> 4;
  int nm = lane & 15;
  int l0 = lt * 16;
  int b0a = wave * 16;
  int b0b = (wave + 4) * 16;
  float dval = Dp[layer * Hsz + h];
  const __bf16* rowA = Ub + (size_t)h * LB + (size_t)(b0a + nm) * Lsz;
  const __bf16* rowB = Ub + (size_t)h * LB + (size_t)(b0b + nm) * Lsz;

  v8f c0 = {}, c1 = {};
  for (int m0 = 0; m0 < Lsz; m0 += 32) {
    __builtin_prefetch(rowA + m0 + 128, 0, 0);   // global_prefetch_b8
    __builtin_prefetch(rowB + m0 + 128, 0, 0);
    v16bf a, bA, bB;
#pragma unroll
    for (int e = 0; e < 16; ++e) {
      int k  = frag_k(e, hi);
      int lr = l0 + nm;                 // A row (M)
      int mc = m0 + k;                  // reduction index
      a[e]  = (mc <= lr) ? sk0[lr - mc] : sk1[mc - lr - 1];
      bA[e] = rowA[m0 + k];
      bB[e] = rowB[m0 + k];
    }
    c0 = wmma_bf16(a, bA, c0);
    c1 = wmma_bf16(a, bB, c1);
  }

  const float* Ufh = U + (size_t)h * LB;
  __bf16* Yh = Yact + (size_t)h * LB;
#pragma unroll
  for (int r = 0; r < 8; ++r) {
    int l = l0 + r + hi * 8;
    {
      size_t idx = (size_t)(b0a + nm) * Lsz + l;
      float v = c0[r] + dval * Ufh[idx];
      Yh[idx] = (__bf16)gelu_tanh(v);
    }
    {
      size_t idx = (size_t)(b0b + nm) * Lsz + l;
      float v = c1[r] + dval * Ufh[idx];
      Yh[idx] = (__bf16)gelu_tanh(v);
    }
  }
}

// ---------------- output linear (H->2H) + GLU + residual ----------------
// One block per 16-column tile jt. The 256x16 bf16 Yact column tile is staged
// into LDS once via the Tensor Data Mover, then shared by 4 waves x 4 o-tiles.
// Each wave computes matching 'a' and 'g' row tiles so GLU fuses in registers.
__global__ void outglu_kernel(const __bf16* __restrict__ W2b,
                              const float* __restrict__ out_b,
                              const __bf16* __restrict__ Yact,
                              const float* __restrict__ U,
                              float* __restrict__ U2,
                              int layer) {
  __shared__ __bf16 sY[Hsz * 16];       // 8 KB: row k=h (256 rows) x 16 cols
  int jt = blockIdx.x;                  // 4096 column tiles
  int wave = threadIdx.x >> 5;
  int lane = threadIdx.x & 31;
  int hi = lane >> 4;
  int nm = lane & 15;

  if (wave == 0) {
    // TDM D#: 2D tile, 16 elements/row (dim0, contiguous), 256 rows (dim1),
    // row stride = LB elements, element size 2 bytes.
    unsigned long long gaddr = (unsigned long long)(Yact + (size_t)jt * 16);
    unsigned int laddr = (unsigned int)(size_t)(void*)sY;   // LDS byte offset
    v4u g0;
    g0[0] = 1u;                                   // count=1, user mode
    g0[1] = laddr;                                // lds_addr
    g0[2] = (unsigned int)gaddr;                  // global_addr[31:0]
    g0[3] = (unsigned int)((gaddr >> 32) & 0x1FFFFFFu) | (2u << 30); // addr[56:32], type=2
    v8i g1;
    g1[0] = (int)(1u << 16);                      // data_size = 1 (2 bytes)
    g1[1] = 0;                                    // tensor_dim0[15:0] of 65536 -> 0
    g1[2] = (int)0x01000001u;                     // tensor_dim0 hi=1 | tensor_dim1(256) lo
    g1[3] = (int)(16u << 16);                     // tile_dim0 = 16
    g1[4] = 256;                                  // tile_dim1 = 256, tile_dim2 = 0
    g1[5] = (int)(unsigned int)LB;                // tensor_dim0_stride = 65536
    g1[6] = 0;
    g1[7] = 0;
    v4i gz4 = {0, 0, 0, 0};
    v8i gz8 = {0, 0, 0, 0, 0, 0, 0, 0};
    __builtin_amdgcn_tensor_load_to_lds(g0, g1, gz4, gz4, gz8, 0);
    __builtin_amdgcn_s_wait_tensorcnt(0);
  }
  __syncthreads();

  const __bf16* Wl = W2b + (size_t)layer * 2 * Hsz * Hsz;
  int jc = jt * 16 + nm;

  v8f ca[4] = {}, cg[4] = {};
  for (int kk = 0; kk < Hsz; kk += 32) {
    v16bf bb;
#pragma unroll
    for (int e = 0; e < 16; ++e)
      bb[e] = sY[(kk + frag_k(e, hi)) * 16 + nm];
#pragma unroll
    for (int i = 0; i < 4; ++i) {
      int ot = wave * 4 + i;
      v16bf a1, a2;
#pragma unroll
      for (int e = 0; e < 16; ++e) {
        int k = frag_k(e, hi);
        a1[e] = Wl[(size_t)(ot * 16 + nm) * Hsz + kk + k];
        a2[e] = Wl[(size_t)(Hsz + ot * 16 + nm) * Hsz + kk + k];
      }
      ca[i] = wmma_bf16(a1, bb, ca[i]);
      cg[i] = wmma_bf16(a2, bb, cg[i]);
    }
  }

#pragma unroll
  for (int i = 0; i < 4; ++i) {
    int ot = wave * 4 + i;
#pragma unroll
    for (int r = 0; r < 8; ++r) {
      int o = ot * 16 + r + hi * 8;
      float av = ca[i][r] + out_b[layer * 2 * Hsz + o];
      float gv = cg[i][r] + out_b[layer * 2 * Hsz + Hsz + o];
      float v  = av * (1.f / (1.f + __expf(-gv))) + U[(size_t)o * LB + jc];
      U2[(size_t)o * LB + jc] = v;
    }
  }
}

// ---------------- post-norm over channels (columns of H x LB) ----------------
__global__ void ln_kernel(const float* __restrict__ U2,
                          const float* __restrict__ g,
                          const float* __restrict__ b,
                          int layer,
                          float* __restrict__ U,
                          __bf16* __restrict__ Ub) {
  int jc = blockIdx.x * blockDim.x + threadIdx.x;   // LB total, coalesced
  float s = 0.f, ss = 0.f;
  for (int h = 0; h < Hsz; ++h) {
    float v = U2[(size_t)h * LB + jc];
    s += v; ss += v * v;
  }
  float mu  = s * (1.f / Hsz);
  float var = ss * (1.f / Hsz) - mu * mu;
  float inv = rsqrtf(var + 1e-5f);
  for (int h = 0; h < Hsz; ++h) {
    float v = (U2[(size_t)h * LB + jc] - mu) * inv * g[layer * Hsz + h] + b[layer * Hsz + h];
    U [(size_t)h * LB + jc] = v;
    Ub[(size_t)h * LB + jc] = (__bf16)v;
  }
}

// ---------------- mean over time, decoder ----------------
__global__ void pool_kernel(const float* __restrict__ U, float* __restrict__ feat) {
  int h = blockIdx.x;        // H
  int b = threadIdx.x;       // B
  const float4* p = (const float4*)(U + (size_t)h * LB + (size_t)b * Lsz);
  float s = 0.f;
  for (int i = 0; i < Lsz / 4; ++i) {
    float4 v = p[i];
    s += v.x + v.y + v.z + v.w;
  }
  feat[h * Bsz + b] = s * (1.f / Lsz);
}

__global__ void dec_kernel(const float* __restrict__ feat,
                           const float* __restrict__ dec_w,
                           const float* __restrict__ dec_b,
                           float* __restrict__ out) {
  int t = threadIdx.x;       // 256 = B*2
  int b = t >> 1, c = t & 1;
  float s = dec_b[c];
  for (int h = 0; h < Hsz; ++h) s += feat[h * Bsz + b] * dec_w[c * Hsz + h];
  out[b * 2 + c] = s;
}

// ---------------- orchestration ----------------
extern "C" void kernel_launch(void* const* d_in, const int* in_sizes, int n_in,
                              void* d_out, int out_size, void* d_ws, size_t ws_size,
                              hipStream_t stream) {
  const float* x          = (const float*)d_in[0];
  const float* enc_w      = (const float*)d_in[1];
  const float* enc_b      = (const float*)d_in[2];
  const float* log_dt     = (const float*)d_in[3];
  const float* log_A_real = (const float*)d_in[4];
  const float* A_imag     = (const float*)d_in[5];
  const float* Cr         = (const float*)d_in[6];
  const float* Ci         = (const float*)d_in[7];
  const float* Dp         = (const float*)d_in[8];
  const float* out_w      = (const float*)d_in[9];
  const float* out_b      = (const float*)d_in[10];
  const float* ln_g       = (const float*)d_in[11];
  const float* ln_b       = (const float*)d_in[12];
  const float* dec_w      = (const float*)d_in[13];
  const float* dec_b      = (const float*)d_in[14];

  const size_t HLB = (size_t)Hsz * LB;              // 16,777,216
  char* ws = (char*)d_ws;
  size_t off = 0;
  auto take = [&](size_t bytes) { char* p = ws + off; off = (off + bytes + 255) & ~(size_t)255; return p; };
  float*  U     = (float*) take(HLB * 4);           // f32 master activations (H,B,L)
  float*  U2    = (float*) take(HLB * 4);           // pre-LN buffer
  __bf16* Ubf   = (__bf16*)take(HLB * 2);           // bf16 GEMM operand copy
  __bf16* Yact  = (__bf16*)take(HLB * 2);           // post-GELU conv output
  float*  k0    = (float*) take((size_t)Hsz * Lsz * 4);
  float*  k1    = (float*) take((size_t)Hsz * Lsz * 4);
  __bf16* Wencb = (__bf16*)take((size_t)Hsz * Hsz * 2);
  __bf16* Woutb = (__bf16*)take((size_t)NLayers * 2 * Hsz * Hsz * 2);
  float*  feat  = (float*) take((size_t)Hsz * Bsz * 4);
  (void)ws_size; (void)in_sizes; (void)n_in; (void)out_size;

  // 1) weights -> bf16 once
  wconv_kernel<<<2048, 256, 0, stream>>>(enc_w, out_w, Wencb, Woutb);

  // 2) encoder GEMM
  encoder_kernel<<<16384, 128, 0, stream>>>(x, Wencb, enc_b, U, Ubf);

  // 3) layers
  for (int i = 0; i < NLayers; ++i) {
    kgen_kernel<<<Hsz, Lsz, 0, stream>>>(log_dt, log_A_real, A_imag, Cr, Ci, i, k0, k1);
    conv_kernel<<<Hsz * (Lsz / 16), 128, 0, stream>>>(Ubf, U, k0, k1, Dp, i, Yact);
    outglu_kernel<<<LB / 16, 128, 0, stream>>>(Woutb, out_b, Yact, U, U2, i);
    ln_kernel<<<LB / 256, 256, 0, stream>>>(U2, ln_g, ln_b, i, U, Ubf);
  }

  // 4) head
  pool_kernel<<<Hsz, Bsz, 0, stream>>>(U, feat);
  dec_kernel<<<1, 256, 0, stream>>>(feat, dec_w, dec_b, (float*)d_out);
}
